// AdversarialLoss_PDD_69776038691189
// MI455X (gfx1250) — matrix operand order repaired and checked by reading, not compiled, not run on hardware
//
#include <hip/hip_runtime.h>
#include <hip/hip_bf16.h>

// ---------------- problem constants ----------------
#define IN_FEATURES 2048
#define N_CLASSES   128
#define BATCH       1024
#define BS          512          // BATCH/2
#define TEMP        2.0f
#define FEATURE_TEMP 2.0f
#define THRESHOLD   0.05f
#define FACTOR      1.0f

typedef __attribute__((ext_vector_type(16))) _Float16 v16h;
typedef __attribute__((ext_vector_type(8)))  float    v8f;

typedef union { v16h v; unsigned int u[8]; } frag16;

// ---------------- workspace layout (bytes) ----------------
#define OFF_FH     ((size_t)0)                         // half[1024*2048]  4 MB
#define OFF_WH     (OFF_FH   + (size_t)BATCH*IN_FEATURES*2)       // half[128*2048]
#define OFF_Y      (OFF_WH   + (size_t)N_CLASSES*IN_FEATURES*2)   // float[1024*128]
#define OFF_S      (OFF_Y    + (size_t)BATCH*N_CLASSES*4)         // float[1024*128]
#define OFF_ST     (OFF_S    + (size_t)BATCH*N_CLASSES*4)         // float[128*1024] (class-major)
#define OFF_NEG    (OFF_ST   + (size_t)BATCH*N_CLASSES*4)         // float[1024]
#define OFF_CONF   (OFF_NEG  + (size_t)BATCH*4)                   // float[512]
#define OFF_PSEUDO (OFF_CONF + (size_t)BS*4)                      // int[512]
#define OFF_PSUM   (OFF_PSEUDO + (size_t)BS*4)                    // float[2048]
#define OFF_PCNT   (OFF_PSUM + (size_t)2048*4)                    // float[2048]

// ---------------- kernel 1: f32 -> f16 convert (f then W) ----------------
__global__ void cvt_f16_kernel(const float* __restrict__ f, const float* __restrict__ W,
                               _Float16* __restrict__ fh, _Float16* __restrict__ Wh) {
    const int i = blockIdx.x * blockDim.x + threadIdx.x;
    const int nf = BATCH * IN_FEATURES;
    const int nw = N_CLASSES * IN_FEATURES;
    if (i < nf) fh[i] = (_Float16)f[i];
    else if (i < nf + nw) Wh[i - nf] = (_Float16)W[i - nf];
}

// ---------------- kernel 2: WMMA GEMM  y = (f @ W^T + b) / FEATURE_TEMP ----------------
// One wave32 per 16x16 output tile; K-loop of 64 x v_wmma_f32_16x16x32_f16.
__global__ void gemm_wmma_kernel(const _Float16* __restrict__ fh,
                                 const _Float16* __restrict__ Wh,
                                 const float* __restrict__ bias,
                                 float* __restrict__ y) {
    const int wave = (blockIdx.x * blockDim.x + threadIdx.x) >> 5;
    const int lane = threadIdx.x & 31;
    const int NT_N = N_CLASSES / 16;            // 8 tiles in N
    const int tm = wave / NT_N;                 // 0..63
    const int tn = wave % NT_N;                 // 0..7
    const int m0 = tm * 16, n0 = tn * 16;
    const int lhalf = lane >> 4;                // 0 or 1
    const int l16   = lane & 15;

    // A: row (m0+l16) of f; B: column (n0+l16) of B == row (n0+l16) of W (contiguous K)
    const unsigned int* __restrict__ arow =
        (const unsigned int*)(fh + (size_t)(m0 + l16) * IN_FEATURES);
    const unsigned int* __restrict__ brow =
        (const unsigned int*)(Wh + (size_t)(n0 + l16) * IN_FEATURES);

    v8f c = {};
    for (int k0 = 0; k0 < IN_FEATURES; k0 += 32) {
        frag16 a, b;
        const unsigned int* ap = arow + (k0 >> 1);   // packed f16 pairs
        const unsigned int* bp = brow + (k0 >> 1);
        // A-matrix 16x32 f16 layout (ISA 7.12.2): lanes0-15 K=0..7/16..23, lanes16-31 +8
        #pragma unroll
        for (int i = 0; i < 4; ++i) {
            a.u[i]     = ap[lhalf * 4 + i];          // K = lhalf*8 + 2i
            a.u[4 + i] = ap[8 + lhalf * 4 + i];      // K = 16 + lhalf*8 + 2i
        }
        // B-matrix 32x16 f16: same K order as A (lanes0-15 K=0..7/16..23, lanes16-31 +8)
        #pragma unroll
        for (int i = 0; i < 4; ++i) {
            b.u[i]     = bp[lhalf * 4 + i];
            b.u[4 + i] = bp[8 + lhalf * 4 + i];
        }
        c = __builtin_amdgcn_wmma_f32_16x16x32_f16(
                /*neg_a=*/false, a.v, /*neg_b=*/false, b.v,
                /*c_mod=*/(short)0, c, /*reuse_a=*/false, /*reuse_b=*/false);
    }
    // C/D 16x16 f32 layout: VGPR i -> M = m0 + 8*lhalf + i, N = n0 + l16
    const int n = n0 + l16;
    const float bn = bias[n];
    #pragma unroll
    for (int i = 0; i < 8; ++i) {
        const int m = m0 + 8 * lhalf + i;
        y[(size_t)m * N_CLASSES + n] = (c[i] + bn) * (1.0f / FEATURE_TEMP);
    }
}

// ---------------- kernel 3: per-row softmax stats ----------------
// block = 128 threads (one per class), grid = 1024 rows.
__global__ void rowstats_kernel(const float* __restrict__ y,
                                float* __restrict__ S,      // [1024][128] row-major
                                float* __restrict__ ST,     // [128][1024] class-major
                                float* __restrict__ negent, // [1024]
                                float* __restrict__ conf_t, // [512]
                                int*   __restrict__ pseudo_t) { // [512]
    const int row = blockIdx.x;
    const int c   = threadIdx.x;
    __shared__ float red[N_CLASSES];
    __shared__ int   redi[N_CLASSES];

    const float v = y[(size_t)row * N_CLASSES + c];

    // max over classes
    red[c] = v; __syncthreads();
    for (int s = N_CLASSES / 2; s > 0; s >>= 1) {
        if (c < s) red[c] = fmaxf(red[c], red[c + s]);
        __syncthreads();
    }
    const float m1 = red[0]; __syncthreads();

    // softmax @ temp 1 (for conf) + first-argmax
    const float e1 = expf(v - m1);
    red[c]  = e1;
    redi[c] = (v == m1) ? c : N_CLASSES;   // first index achieving the max
    __syncthreads();
    for (int s = N_CLASSES / 2; s > 0; s >>= 1) {
        if (c < s) { red[c] += red[c + s]; redi[c] = min(redi[c], redi[c + s]); }
        __syncthreads();
    }
    const float sum1 = red[0];
    const int   amax = redi[0];
    __syncthreads();

    // S = softmax(v / TEMP); max scales with positive factor
    const float l2 = (v - m1) * (1.0f / TEMP);
    const float e2 = expf(l2);
    red[c] = e2; __syncthreads();
    for (int s = N_CLASSES / 2; s > 0; s >>= 1) {
        if (c < s) red[c] += red[c + s];
        __syncthreads();
    }
    const float sum2 = red[0]; __syncthreads();

    const float sv = e2 / sum2;
    S [(size_t)row * N_CLASSES + c] = sv;
    ST[(size_t)c * BATCH + row]     = sv;

    // negative entropy contribution: s*log(s) = s*(l2 - log(sum2))
    red[c] = sv * (l2 - logf(sum2)); __syncthreads();
    for (int s = N_CLASSES / 2; s > 0; s >>= 1) {
        if (c < s) red[c] += red[c + s];
        __syncthreads();
    }
    if (c == 0) {
        negent[row] = red[0];
        if (row >= BS) {
            conf_t[row - BS]   = 1.0f / sum1;  // exp(m1-m1)/sum1
            pseudo_t[row - BS] = amax;
        }
    }
}

// ---------------- kernel 4: masked JS with deterministic per-block partials ----------------
// grid = (2, 512, 2): x = j-chunk, y = i, z = 0 (st) / 1 (ss). block = 256.
__global__ void js_kernel(const float* __restrict__ S,
                          const float* __restrict__ ST,
                          const float* __restrict__ negent,
                          const float* __restrict__ conf_t,
                          const int*   __restrict__ pseudo_t,
                          const int*   __restrict__ labels,
                          float* __restrict__ psum,
                          float* __restrict__ pcnt) {
    const int which = blockIdx.z;                       // 0: Ss-vs-St, 1: Ss-vs-Ss
    const int i = blockIdx.y;                           // source row 0..511
    const int j = blockIdx.x * blockDim.x + threadIdx.x;// 0..511

    __shared__ float sP[N_CLASSES];
    if (threadIdx.x < N_CLASSES) sP[threadIdx.x] = S[(size_t)i * N_CLASSES + threadIdx.x];
    __syncthreads();

    const int labi = labels[i];
    bool mask; int qrow;
    if (which == 0) { mask = (labi == pseudo_t[j]) && (conf_t[j] >= THRESHOLD); qrow = BS + j; }
    else            { mask = (j > i) && (labi == labels[j]);                    qrow = j;      }

    float js = 0.0f, cnt = 0.0f;
    if (mask) {
        float acc = 0.0f;
        #pragma unroll 4
        for (int cc = 0; cc < N_CLASSES; ++cc) {
            const float p  = sP[cc];
            const float q  = ST[(size_t)cc * BATCH + qrow];   // coalesced in j
            const float pq = p + q;
            acc += pq * logf(0.5f * pq);                      // (P+Q)*log(M)
        }
        js  = 0.5f * (negent[i] + negent[qrow] - acc);
        cnt = 1.0f;
    }

    __shared__ float rs[256], rc[256];
    rs[threadIdx.x] = js; rc[threadIdx.x] = cnt; __syncthreads();
    for (int s = 128; s > 0; s >>= 1) {
        if (threadIdx.x < s) { rs[threadIdx.x] += rs[threadIdx.x + s];
                               rc[threadIdx.x] += rc[threadIdx.x + s]; }
        __syncthreads();
    }
    if (threadIdx.x == 0) {
        const int idx = which * 1024 + i * 2 + blockIdx.x;    // every slot written -> no init needed
        psum[idx] = rs[0];
        pcnt[idx] = rc[0];
    }
}

// ---------------- kernel 5: final deterministic reduction ----------------
__global__ void final_kernel(const float* __restrict__ psum,
                             const float* __restrict__ pcnt,
                             float* __restrict__ out) {
    __shared__ float s_st[256], c_st[256], s_ss[256], c_ss[256];
    const int t = threadIdx.x;
    float a = 0.f, b = 0.f, c = 0.f, d = 0.f;
    for (int idx = t; idx < 1024; idx += 256) {
        a += psum[idx];        b += pcnt[idx];         // st
        c += psum[1024 + idx]; d += pcnt[1024 + idx];  // ss
    }
    s_st[t] = a; c_st[t] = b; s_ss[t] = c; c_ss[t] = d; __syncthreads();
    for (int s = 128; s > 0; s >>= 1) {
        if (t < s) { s_st[t] += s_st[t+s]; c_st[t] += c_st[t+s];
                     s_ss[t] += s_ss[t+s]; c_ss[t] += c_ss[t+s]; }
        __syncthreads();
    }
    if (t == 0) {
        const float mst = (c_st[0] > 0.f) ? s_st[0] / c_st[0] : 0.f;
        const float mss = (c_ss[0] > 0.f) ? s_ss[0] / c_ss[0] : 0.f;
        out[0] = TEMP * TEMP * (mss + mst) * FACTOR;
        out[1] = 0.0f;
    }
}

// ---------------- launch ----------------
extern "C" void kernel_launch(void* const* d_in, const int* in_sizes, int n_in,
                              void* d_out, int out_size, void* d_ws, size_t ws_size,
                              hipStream_t stream) {
    (void)in_sizes; (void)n_in; (void)out_size; (void)ws_size;
    const float* f      = (const float*)d_in[0];
    const float* W      = (const float*)d_in[1];
    const float* bias   = (const float*)d_in[2];
    const int*   labels = (const int*)d_in[3];

    char* ws = (char*)d_ws;
    _Float16* fh     = (_Float16*)(ws + OFF_FH);
    _Float16* Wh     = (_Float16*)(ws + OFF_WH);
    float* y         = (float*)(ws + OFF_Y);
    float* S         = (float*)(ws + OFF_S);
    float* ST        = (float*)(ws + OFF_ST);
    float* negent    = (float*)(ws + OFF_NEG);
    float* conf_t    = (float*)(ws + OFF_CONF);
    int*   pseudo_t  = (int*)(ws + OFF_PSEUDO);
    float* psum      = (float*)(ws + OFF_PSUM);
    float* pcnt      = (float*)(ws + OFF_PCNT);
    float* out       = (float*)d_out;

    // 1) convert inputs to f16
    {
        const int total = BATCH * IN_FEATURES + N_CLASSES * IN_FEATURES;
        cvt_f16_kernel<<<(total + 255) / 256, 256, 0, stream>>>(f, W, fh, Wh);
    }
    // 2) WMMA GEMM: 64x8 = 512 tiles, one wave each -> 64 blocks x 256 threads
    {
        const int waves = (BATCH / 16) * (N_CLASSES / 16);   // 512
        gemm_wmma_kernel<<<waves * 32 / 256, 256, 0, stream>>>(fh, Wh, bias, y);
    }
    // 3) per-row softmax stats
    rowstats_kernel<<<BATCH, N_CLASSES, 0, stream>>>(y, S, ST, negent, conf_t, pseudo_t);
    // 4) masked JS partials
    {
        dim3 grid(BS / 256, BS, 2);
        js_kernel<<<grid, 256, 0, stream>>>(S, ST, negent, conf_t, pseudo_t, labels, psum, pcnt);
    }
    // 5) final reduction
    final_kernel<<<1, 256, 0, stream>>>(psum, pcnt, out);
}
